// Attention_26757646254143
// MI455X (gfx1250) — matrix operand order repaired and checked by reading, not compiled, not run on hardware
//
#include <hip/hip_runtime.h>
#include <hip/hip_bf16.h>
#include <math.h>

// Problem constants (B=32, T=2048, H=1024)
#define BATCH 32
#define TLEN  2048
#define HDIM  1024
#define MTOT  (BATCH * TLEN)   // 65536 rows of X

// GEMM tiling
#define MT  128
#define NT  128
#define KT  32
#define LDP 40                 // padded LDS k-stride (bf16 elems) to avoid bank conflicts

typedef __attribute__((ext_vector_type(16))) __bf16 v16bf;
typedef __attribute__((ext_vector_type(8)))  float  v8f;
typedef int v4i __attribute__((vector_size(16)));   // matches async-LDS builtin param

// ---- CDNA5 feature gates ---------------------------------------------------
#if defined(__gfx1250__) && __has_builtin(__builtin_amdgcn_global_load_async_to_lds_b128)
#define HAVE_ASYNC_LDS 1
#else
#define HAVE_ASYNC_LDS 0
#endif

#if HAVE_ASYNC_LDS
#if __has_builtin(__builtin_amdgcn_s_wait_asynccnt)
#define ASYNC_WAIT() __builtin_amdgcn_s_wait_asynccnt(0)
#else
#define ASYNC_WAIT() asm volatile("s_wait_asynccnt 0x0" ::: "memory")
#endif
#else
#define ASYNC_WAIT() ((void)0)
#endif

#if defined(__gfx1250__) && __has_builtin(__builtin_amdgcn_tanhf)
#define TANHF(x) __builtin_amdgcn_tanhf(x)
#else
#define TANHF(x) tanhf(x)
#endif

// ---------------------------------------------------------------------------
// Kernel 0: Wt[n][k] = bf16(W[k][n]) ; zero logits scratch
// ---------------------------------------------------------------------------
__global__ __launch_bounds__(256) void prep_kernel(const float* __restrict__ W,
                                                   __bf16* __restrict__ Wt,
                                                   float* __restrict__ logits) {
  const int n = blockIdx.x;  // 0..HDIM-1
  for (int k = threadIdx.x; k < HDIM; k += blockDim.x) {
    Wt[(size_t)n * HDIM + k] = (__bf16)W[(size_t)k * HDIM + n];
  }
  const int g = blockIdx.x * blockDim.x + threadIdx.x;
  if (g < MTOT) logits[g] = 0.0f;
}

// ---------------------------------------------------------------------------
// Kernel 1: per block: tile [MT x NT] of score = tanh(X*W); fused dot with c,
// atomicAdd partial logit sums per row. bf16 WMMA, f32 accumulate.
// ---------------------------------------------------------------------------
__global__ __launch_bounds__(256) void gemm_tanh_logits(const float* __restrict__ x,
                                                        const __bf16* __restrict__ Wt,
                                                        const float* __restrict__ c,
                                                        float* __restrict__ logits) {
  __shared__ __bf16 xs[2][MT][LDP];   // A tiles (row-major, K contiguous)
  __shared__ __bf16 wsh[2][NT][LDP];  // B tiles (col of W, K contiguous)
  __shared__ float  cs[NT];

  const int tid   = threadIdx.x;
  const int lane  = tid & 31;
  const int wave  = tid >> 5;        // 0..7, owns rows [wave*16, wave*16+16)
  const int l16   = lane & 15;
  const int hi    = (lane >> 4) & 1; // upper 16 lanes of the wave

  const int nBase = blockIdx.x * NT;
  const int mBase = blockIdx.y * MT;

  if (tid < NT) cs[tid] = c[nBase + tid];

  // --- tile loaders -------------------------------------------------------
  auto loadX = [&](int buf, int kstep) {
    // 128 rows x 32 k of f32 -> bf16. 256 threads: 2 threads per row.
    const int r  = tid >> 1;
    const int kh = (tid & 1) * 16;
    const float* src = x + (size_t)(mBase + r) * HDIM + kstep * KT + kh;
    float tmp[16];
    *(float4*)(tmp + 0)  = ((const float4*)src)[0];
    *(float4*)(tmp + 4)  = ((const float4*)src)[1];
    *(float4*)(tmp + 8)  = ((const float4*)src)[2];
    *(float4*)(tmp + 12) = ((const float4*)src)[3];
    __bf16* dst = &xs[buf][r][kh];
#pragma unroll
    for (int i = 0; i < 16; ++i) dst[i] = (__bf16)tmp[i];
  };

  auto loadW = [&](int buf, int kstep) {
    // 128 cols x 32 k of bf16: pure copy global -> LDS.
    const int n  = tid >> 1;
    const int kh = (tid & 1) * 16;
    const __bf16* src = Wt + (size_t)(nBase + n) * HDIM + kstep * KT + kh;
    __bf16* dst = &wsh[buf][n][kh];
#if HAVE_ASYNC_LDS
    // CDNA5 async DMA: cache -> LDS without touching VGPRs (ASYNCcnt tracked).
    __builtin_amdgcn_global_load_async_to_lds_b128(
        (__attribute__((address_space(1))) v4i*)src,
        (__attribute__((address_space(3))) v4i*)dst, 0, 0);
    __builtin_amdgcn_global_load_async_to_lds_b128(
        (__attribute__((address_space(1))) v4i*)(src + 8),
        (__attribute__((address_space(3))) v4i*)(dst + 8), 0, 0);
#else
    const uint4* s4 = (const uint4*)src;
    uint4* d4 = (uint4*)dst;
    d4[0] = s4[0];
    d4[1] = s4[1];
#endif
  };

  // --- main loop ----------------------------------------------------------
  v8f acc[8];
#pragma unroll
  for (int i = 0; i < 8; ++i) acc[i] = (v8f){};

  loadX(0, 0);
  loadW(0, 0);
  ASYNC_WAIT();
  __syncthreads();

  const int KSTEPS = HDIM / KT;  // 32
  for (int ks = 0; ks < KSTEPS; ++ks) {
    const int buf = ks & 1;
    if (ks + 1 < KSTEPS) {
      loadX(buf ^ 1, ks + 1);
      loadW(buf ^ 1, ks + 1);
    }

    // A fragment: 16x32 bf16: lanes 0-15 -> K[0..15];
    // lanes 16-31 -> K[16..31]; lane row = l16.
    v16bf a;
    {
      const __bf16* xr = &xs[buf][wave * 16 + l16][hi * 16];
#pragma unroll
      for (int i = 0; i < 8; ++i) {
        a[i]     = xr[i];
        a[8 + i] = xr[8 + i];
      }
    }

#pragma unroll
    for (int ns = 0; ns < 8; ++ns) {
      // B fragment: 32x16 bf16: lane column = l16, lanes 0-15 hold K[0..15],
      // lanes 16-31 hold K[16..31] (K contiguous per lane).
      v16bf bfr;
      const __bf16* wr = &wsh[buf][ns * 16 + l16][hi * 16];
#pragma unroll
      for (int i = 0; i < 16; ++i) bfr[i] = wr[i];

      acc[ns] = __builtin_amdgcn_wmma_f32_16x16x32_bf16(
          false, a, false, bfr, (short)0, acc[ns], false, false);
    }
    ASYNC_WAIT();     // async fills of buf^1 must complete before next read
    __syncthreads();
  }

  // --- fused epilogue: part[r] = sum_n tanh(acc) * c[n] -------------------
  // C/D layout: VGPR r, lanes 0-15 -> (row r, col l16); lanes 16-31 -> (row r+8, col l16)
  float part[8];
#pragma unroll
  for (int r = 0; r < 8; ++r) part[r] = 0.0f;

#pragma unroll
  for (int ns = 0; ns < 8; ++ns) {
    const float cv = cs[ns * 16 + l16];
#pragma unroll
    for (int r = 0; r < 8; ++r) part[r] += TANHF(acc[ns][r]) * cv;
  }

#pragma unroll
  for (int r = 0; r < 8; ++r) {
    float v = part[r];
    v += __shfl_xor(v, 1, 32);
    v += __shfl_xor(v, 2, 32);
    v += __shfl_xor(v, 4, 32);
    v += __shfl_xor(v, 8, 32);  // reduces within each 16-lane half
    if (l16 == 0) {
      const int row = mBase + wave * 16 + hi * 8 + r;
      atomicAdd(&logits[row], v);
    }
  }
}

// ---------------------------------------------------------------------------
// Kernel 2: softmax over T per batch; writes attn to d_out; zeroes context.
// d_out layout: [context (B*H)] [attn (B*T)]
// ---------------------------------------------------------------------------
__global__ __launch_bounds__(256) void softmax_kernel(const float* __restrict__ logits,
                                                      float* __restrict__ out) {
  __shared__ float red[256];
  const int b   = blockIdx.x;
  const int tid = threadIdx.x;
  const float* l = logits + (size_t)b * TLEN;
  float* attn = out + (size_t)BATCH * HDIM + (size_t)b * TLEN;
  float* ctx  = out + (size_t)b * HDIM;

  for (int h = tid; h < HDIM; h += 256) ctx[h] = 0.0f;

  float mx = -INFINITY;
  for (int t = tid; t < TLEN; t += 256) mx = fmaxf(mx, l[t]);
  red[tid] = mx;
  __syncthreads();
  for (int s = 128; s > 0; s >>= 1) {
    if (tid < s) red[tid] = fmaxf(red[tid], red[tid + s]);
    __syncthreads();
  }
  mx = red[0];
  __syncthreads();

  float sum = 0.0f;
  for (int t = tid; t < TLEN; t += 256) sum += __expf(l[t] - mx);
  red[tid] = sum;
  __syncthreads();
  for (int s = 128; s > 0; s >>= 1) {
    if (tid < s) red[tid] += red[tid + s];
    __syncthreads();
  }
  const float inv = 1.0f / red[0];

  for (int t = tid; t < TLEN; t += 256) attn[t] = __expf(l[t] - mx) * inv;
}

// ---------------------------------------------------------------------------
// Kernel 3: context[b][h] += sum_t x[b][t][h] * attn[b][t]  (T-split, atomic)
// grid: (H/256, T/256, B)
// ---------------------------------------------------------------------------
__global__ __launch_bounds__(256) void context_kernel(const float* __restrict__ x,
                                                      float* __restrict__ out) {
  __shared__ float sa[256];
  const int b  = blockIdx.z;
  const int t0 = blockIdx.y * 256;
  const int h  = blockIdx.x * 256 + threadIdx.x;

  const float* attn = out + (size_t)BATCH * HDIM + (size_t)b * TLEN;
  sa[threadIdx.x] = attn[t0 + threadIdx.x];
  __syncthreads();

  const float* xb = x + ((size_t)b * TLEN + t0) * HDIM + h;
  float s = 0.0f;
#pragma unroll 4
  for (int t = 0; t < 256; ++t) s += xb[(size_t)t * HDIM] * sa[t];

  atomicAdd(&out[(size_t)b * HDIM + h], s);
}

// ---------------------------------------------------------------------------
extern "C" void kernel_launch(void* const* d_in, const int* in_sizes, int n_in,
                              void* d_out, int out_size, void* d_ws, size_t ws_size,
                              hipStream_t stream) {
  const float* x = (const float*)d_in[0];  // [B,T,H] f32
  const float* W = (const float*)d_in[1];  // [H,H]   f32
  const float* c = (const float*)d_in[2];  // [H,1]   f32
  float* out = (float*)d_out;              // [B*H context][B*T attn]

  __bf16* Wt     = (__bf16*)d_ws;                                                 // 2 MB
  float*  logits = (float*)((char*)d_ws + (size_t)HDIM * HDIM * sizeof(__bf16));  // 256 KB

  // 1) transpose+convert W, zero logits
  prep_kernel<<<HDIM, 256, 0, stream>>>(W, Wt, logits);

  // 2) GEMM + tanh + dot(c) -> logits
  dim3 gA(HDIM / NT, MTOT / MT);  // (8, 512)
  gemm_tanh_logits<<<gA, 256, 0, stream>>>(x, Wt, c, logits);

  // 3) softmax over T -> attn; zero context
  softmax_kernel<<<BATCH, 256, 0, stream>>>(logits, out);

  // 4) weighted sum over T -> context
  dim3 gC(HDIM / 256, TLEN / 256, BATCH);  // (4, 8, 32)
  context_kernel<<<gC, 256, 0, stream>>>(x, out);
}
